// RQ_VAE_17403207483788
// MI455X (gfx1250) — compile-verified, matrix operand order checked by reading, not run on hardware
//
#include <hip/hip_runtime.h>
#include <math.h>

// ---------------- problem constants ----------------
constexpr int IN_DIM = 768;
constexpr int LATENT = 64;
constexpr int KCB    = 2048;   // codebook size
constexpr int LQ     = 3;      // residual quantization layers
constexpr float BETA = 0.25f;

// ---------------- CDNA5 WMMA types ----------------
typedef __attribute__((ext_vector_type(4)))  __bf16 v4bf;
typedef __attribute__((ext_vector_type(8)))  __bf16 v8bf;
typedef __attribute__((ext_vector_type(16))) __bf16 v16bf;
typedef __attribute__((ext_vector_type(8)))  float  v8f;
typedef __attribute__((ext_vector_type(4)))  unsigned int v4u;
typedef __attribute__((ext_vector_type(8)))  int v8i;
typedef __attribute__((ext_vector_type(4)))  int v4i;

// Padded LDS row strides (bf16 elements). 40*2B=80B and 72*2B=144B give
// conflict-free 16B ds_read_b128 fragment loads across the 16 row-lanes.
constexpr int LDA = 40;   // for 32-wide K tiles (32 elem row + 8 pad)
constexpr int LDV = 72;   // for 64-wide K tiles (64 elem row + 8 pad)

#if __has_builtin(__builtin_amdgcn_tensor_load_to_lds) && \
    __has_builtin(__builtin_amdgcn_s_wait_tensorcnt)
#define HAVE_TDM 1
#else
#define HAVE_TDM 0
#endif

#if HAVE_TDM
// Tensor Data Mover: async DMA of a 2D bf16 tile (tile_d0 elems x tile_d1 rows)
// from global into LDS, inserting pad_amount DWORDs after every
// 2^(pad_interval+1) DWORDs so the LDS row stride matches LDA/LDV.
// 6-arg builtin form (clang-23 / therock headers): g0, g1, g2, g3, g4, cpol.
__device__ __forceinline__ void tdm_load_2d_bf16(
    unsigned lds_addr, const __bf16* gptr,
    unsigned tile_d0, unsigned tile_d1,
    unsigned tensor_d0, unsigned tensor_d1, unsigned stride_d0,
    unsigned pad_interval_code, unsigned pad_amount_code) {
  const unsigned long long ga = (unsigned long long)(uintptr_t)gptr;
  v4u g0;
  g0.x = 1u;                                          // count=1 (valid user D#)
  g0.y = lds_addr;                                    // lds_addr (bytes)
  g0.z = (unsigned)(ga & 0xFFFFFFFFu);                // global_addr[31:0]
  g0.w = (unsigned)((ga >> 32) & 0x01FFFFFFu)         // global_addr[56:32]
       | (2u << 30);                                  // type = 2 ("image")
  v8i g1;
  g1[0] = (int)((1u << 16)                            // data_size = 2 bytes
              | (1u << 20)                            // pad_enable
              | (pad_interval_code << 22)
              | (pad_amount_code << 25));             // workgroup_mask = 0
  g1[1] = (int)((tensor_d0 & 0xFFFFu) << 16);         // [63:48] tensor_dim0 lo16
  g1[2] = (int)(((tensor_d0 >> 16) & 0xFFFFu)         // [79:64] tensor_dim0 hi16
              | ((tensor_d1 & 0xFFFFu) << 16));       // [95:80] tensor_dim1 lo16
  g1[3] = (int)(((tensor_d1 >> 16) & 0xFFFFu)         // [111:96] tensor_dim1 hi16
              | ((tile_d0 & 0xFFFFu) << 16));         // [127:112] tile_dim0
  g1[4] = (int)(tile_d1 & 0xFFFFu);                   // [143:128] tile_dim1 (dim2=0)
  g1[5] = (int)stride_d0;                             // [191:160] dim0 stride lo32
  g1[6] = 0;                                          // stride hi16 / dim1 stride
  g1[7] = 0;
  const v4i z4 = {0, 0, 0, 0};
  const v8i z8 = {0, 0, 0, 0, 0, 0, 0, 0};
  __builtin_amdgcn_tensor_load_to_lds(g0, g1, z4, z4, z8, 0);
}
#endif

// Build one wave32 bf16 fragment (16 elements / 8 VGPRs) for the 16x16x32
// WMMA from an LDS row. Per ISA layout: lanes 0-15 hold K{0..7,16..23},
// lanes 16-31 hold K{8..15,24..31} -> two contiguous 16B chunks per lane.
__device__ __forceinline__ v16bf lds_frag(const __bf16* rowk, int hi) {
  v8bf lo = *(const v8bf*)(rowk + hi * 8);
  v8bf hh = *(const v8bf*)(rowk + 16 + hi * 8);
  return __builtin_shufflevector(lo, hh, 0,1,2,3,4,5,6,7,8,9,10,11,12,13,14,15);
}

__device__ __forceinline__ void cvt_store4(__bf16* dst, float4 v) {
  v4bf p = { (__bf16)v.x, (__bf16)v.y, (__bf16)v.z, (__bf16)v.w };
  *(v4bf*)dst = p;
}

// ---------------- f32 -> bf16 pre-conversion (weights / codebook) ----------------
__global__ __launch_bounds__(256)
void cvt_bf16_kernel(const float* __restrict__ src, __bf16* __restrict__ dst, int n4) {
  const int i = blockIdx.x * 256 + threadIdx.x;   // one float4 per thread
  if (i < n4) cvt_store4(dst + 4 * (size_t)i, ((const float4*)src)[i]);
}

// =====================================================================
// GEMM: C[M,N] = A[M,K] * W[N,K]^T + bias[N]  (optional ReLU)
// A: f32 (converted to bf16 on the fly), W: pre-converted bf16 (TDM-staged)
// block = 256 threads (8 waves), tile 128(M) x 64(N), BK = 32
// =====================================================================
__global__ __launch_bounds__(256)
void gemm_bias_kernel(const float* __restrict__ A, const __bf16* __restrict__ Wb,
                      const float* __restrict__ bias, float* __restrict__ C,
                      int M, int N, int K, int relu) {
  __shared__ __bf16 As[128 * LDA];
  __shared__ __bf16 Ws[64 * LDA];
  const int tid  = threadIdx.x;
  const int wave = tid >> 5;
  const int lane = tid & 31;
  const int hi   = lane >> 4;
  const int lc   = lane & 15;
  const int blockM = blockIdx.x * 128;
  const int blockN = blockIdx.y * 64;

  const v8f vzero = {0.f,0.f,0.f,0.f,0.f,0.f,0.f,0.f};
  v8f acc[4] = {vzero, vzero, vzero, vzero};

  const int ar = tid >> 1, ah = tid & 1;   // A stage: row, 16-col half
  const float* aptr = A + (size_t)(blockM + ar) * K + ah * 16;
#if !HAVE_TDM
  const int wr = tid >> 2, wq = tid & 3;   // fallback W stage: row, 8-col chunk
#endif

  for (int k0 = 0; k0 < K; k0 += 32) {
#pragma unroll
    for (int j = 0; j < 4; ++j)
      cvt_store4(&As[ar * LDA + ah * 16 + 4 * j], ((const float4*)(aptr + k0))[j]);
    if (k0 + 32 < K) __builtin_prefetch(aptr + k0 + 32, 0, 3);
#if HAVE_TDM
    if (wave == 0) {   // TDM: 64 rows x 32 bf16 (64B) + 4-DWORD pad -> 80B stride
      tdm_load_2d_bf16((unsigned)(uintptr_t)(void*)Ws,
                       Wb + (size_t)blockN * K + k0,
                       /*tile*/ 32, 64,
                       /*tensor*/ (unsigned)K, (unsigned)N, (unsigned)K,
                       /*pad: 16 DW interval*/ 3, /*4 DW amount*/ 3);
      __builtin_amdgcn_s_wait_tensorcnt(0);
    }
#else
    {
      const __bf16* src = Wb + (size_t)(blockN + wr) * K + k0 + wq * 8;
      *(v8bf*)&Ws[wr * LDA + wq * 8] = *(const v8bf*)src;
    }
#endif
    __syncthreads();
    // hoist all fragments, then issue the 4 WMMAs back-to-back
    v16bf bfr[4];
#pragma unroll
    for (int t = 0; t < 4; ++t) bfr[t] = lds_frag(&Ws[(t * 16 + lc) * LDA], hi);
    v16bf af = lds_frag(&As[(wave * 16 + lc) * LDA], hi);
#pragma unroll
    for (int t = 0; t < 4; ++t)
      acc[t] = __builtin_amdgcn_wmma_f32_16x16x32_bf16(false, af, false, bfr[t],
                                                       (short)0, acc[t], false, false);
    __syncthreads();
  }

#pragma unroll
  for (int t = 0; t < 4; ++t) {
    const int col = blockN + t * 16 + lc;
    const float bv = bias[col];
#pragma unroll
    for (int i = 0; i < 8; ++i) {   // D layout: m = i + 8*hi, n = lane&15
      const int row = blockM + wave * 16 + i + hi * 8;
      float v = acc[t][i] + bv;
      if (relu) v = fmaxf(v, 0.0f);
      C[(size_t)row * N + col] = v;
    }
  }
}

// =====================================================================
// Fused VQ argmin: ids[b] = argmin_k ( ||c_k||^2 - 2 * <res_b, c_k> )
// 128 rows/block, bf16 codebook swept in 64-row chunks (TDM-staged, K=64).
// =====================================================================
__global__ __launch_bounds__(256)
void vq_argmin_kernel(const float* __restrict__ res, const __bf16* __restrict__ cbb,
                      const float* __restrict__ cnorm, int* __restrict__ ids) {
  __shared__ __bf16 As[128 * LDV];
  __shared__ __bf16 Ws[64 * LDV];
  __shared__ float  cn[64];
  const int tid  = threadIdx.x;
  const int wave = tid >> 5;
  const int lane = tid & 31;
  const int hi   = lane >> 4;
  const int lc   = lane & 15;
  const int blockM = blockIdx.x * 128;

  { // stage residual tile once (128 x 64 f32 -> bf16)
    const int r = tid >> 1, h = tid & 1;
    const float* src = res + (size_t)(blockM + r) * 64 + h * 32;
#pragma unroll
    for (int j = 0; j < 8; ++j)
      cvt_store4(&As[r * LDV + h * 32 + 4 * j], ((const float4*)src)[j]);
  }

  float best[8];
  int   bidx[8];
#pragma unroll
  for (int i = 0; i < 8; ++i) { best[i] = 3.0e38f; bidx[i] = 0; }

  const v8f vzero = {0.f,0.f,0.f,0.f,0.f,0.f,0.f,0.f};
  for (int c0 = 0; c0 < KCB; c0 += 64) {
    __syncthreads();                         // As ready / Ws readers done
#if HAVE_TDM
    if (wave == 0) {   // TDM: 64 rows x 64 bf16 (128B) + 4-DWORD pad -> 144B stride
      tdm_load_2d_bf16((unsigned)(uintptr_t)(void*)Ws,
                       cbb + (size_t)c0 * 64,
                       /*tile*/ 64, 64,
                       /*tensor*/ 64u, (unsigned)KCB, 64u,
                       /*pad: 32 DW interval*/ 4, /*4 DW amount*/ 3);
      __builtin_amdgcn_s_wait_tensorcnt(0);
    }
#else
    {
      const int cr = tid >> 2, cq = tid & 3;
      const __bf16* src = cbb + (size_t)(c0 + cr) * 64 + cq * 16;
      *(v8bf*)&Ws[cr * LDV + cq * 16]     = *(const v8bf*)src;
      *(v8bf*)&Ws[cr * LDV + cq * 16 + 8] = *(const v8bf*)(src + 8);
    }
#endif
    if (tid < 64) cn[tid] = cnorm[c0 + tid];
    __syncthreads();
    v8f acc[4] = {vzero, vzero, vzero, vzero};
#pragma unroll
    for (int k0 = 0; k0 < 64; k0 += 32) {
      v16bf bfr[4];
#pragma unroll
      for (int t = 0; t < 4; ++t) bfr[t] = lds_frag(&Ws[(t * 16 + lc) * LDV + k0], hi);
      v16bf af = lds_frag(&As[(wave * 16 + lc) * LDV + k0], hi);
#pragma unroll
      for (int t = 0; t < 4; ++t)
        acc[t] = __builtin_amdgcn_wmma_f32_16x16x32_bf16(false, af, false, bfr[t],
                                                         (short)0, acc[t], false, false);
    }
#pragma unroll
    for (int t = 0; t < 4; ++t) {
      const float c = cn[t * 16 + lc];
      const int   n = c0 + t * 16 + lc;
#pragma unroll
      for (int i = 0; i < 8; ++i) {
        const float s = c - 2.0f * acc[t][i];
        if (s < best[i]) { best[i] = s; bidx[i] = n; }
      }
    }
  }

  // reduce argmin across the 16 lanes holding partials of each row
#pragma unroll
  for (int i = 0; i < 8; ++i) {
    float v = best[i];
    int  ix = bidx[i];
#pragma unroll
    for (int m = 8; m >= 1; m >>= 1) {
      const float ov = __shfl_xor(v, m, 32);
      const int   oi = __shfl_xor(ix, m, 32);
      if (ov < v || (ov == v && oi < ix)) { v = ov; ix = oi; }
    }
    if (lc == 0) ids[blockM + wave * 16 + i + hi * 8] = ix;
  }
}

// ---------------- row squared-norms of the projected codebook ----------------
__global__ __launch_bounds__(256)
void rownorm2_kernel(const float* __restrict__ cb, float* __restrict__ cnorm, int rows) {
  const int w    = (int)((blockIdx.x * blockDim.x + threadIdx.x) >> 5); // wave per row
  const int lane = threadIdx.x & 31;
  if (w >= rows) return;
  const float* r = cb + (size_t)w * 64;
  const float a = r[lane], b = r[lane + 32];
  float s = a * a + b * b;
#pragma unroll
  for (int m = 16; m >= 1; m >>= 1) s += __shfl_xor(s, m, 32);
  if (lane == 0) cnorm[w] = s;
}

// ---------------- residual update / commit loss / embedding norms ----------------
__global__ __launch_bounds__(256)
void vq_update_kernel(const float* __restrict__ cb_l, const int* __restrict__ ids_l,
                      float* __restrict__ res, float* __restrict__ esum,
                      float* __restrict__ qloss, float* __restrict__ out_norm,
                      int l, int first, int Bsz) {
  const int b    = (int)((blockIdx.x * blockDim.x + threadIdx.x) >> 5); // wave per row
  const int lane = threadIdx.x & 31;
  if (b >= Bsz) return;
  const int id = ids_l[b];
  const float* e = cb_l + (size_t)id * 64;
  float* rp = res + (size_t)b * 64;
  const float e0 = e[lane], e1 = e[lane + 32];
  const float r0 = rp[lane], r1 = rp[lane + 32];
  const float d0 = e0 - r0, d1 = e1 - r1;
  float sd = d0 * d0 + d1 * d1;
  float se = e0 * e0 + e1 * e1;
#pragma unroll
  for (int m = 16; m >= 1; m >>= 1) { sd += __shfl_xor(sd, m, 32); se += __shfl_xor(se, m, 32); }
  if (lane == 0) {
    const float q = (1.0f + BETA) * sd;            // sg() terms equal in fwd
    qloss[b] = first ? q : (qloss[b] + q);
    out_norm[(size_t)b * LQ + l] = sqrtf(se);      // embs_norm[b, l]
  }
  float* es = esum + (size_t)b * 64;
  if (first) { es[lane] = e0; es[lane + 32] = e1; }
  else       { es[lane] += e0; es[lane + 32] += e1; }
  rp[lane] = r0 - e0; rp[lane + 32] = r1 - e1;     // res -= emb
}

// ---------------- row L2-normalize + reconstruction loss ----------------
__global__ __launch_bounds__(256)
void norm_recon_kernel(const float* __restrict__ xh, const float* __restrict__ x,
                       float* __restrict__ recon_acc) {
  __shared__ float red[8];
  __shared__ float tot;
  const int b    = blockIdx.x;
  const int tid  = threadIdx.x;
  const int wave = tid >> 5;
  const int lane = tid & 31;
  const float* row = xh + (size_t)b * IN_DIM;
  const float* xr  = x  + (size_t)b * IN_DIM;
  float v[3];
  float ss = 0.f;
#pragma unroll
  for (int j = 0; j < 3; ++j) { v[j] = row[tid + j * 256]; ss += v[j] * v[j]; }
#pragma unroll
  for (int m = 16; m >= 1; m >>= 1) ss += __shfl_xor(ss, m, 32);
  if (lane == 0) red[wave] = ss;
  __syncthreads();
  if (tid == 0) {
    float t = 0.f;
#pragma unroll
    for (int k = 0; k < 8; ++k) t += red[k];
    tot = fmaxf(sqrtf(t), 1e-12f);
  }
  __syncthreads();
  const float inv = 1.0f / tot;
  float dd = 0.f;
#pragma unroll
  for (int j = 0; j < 3; ++j) { const float d = v[j] * inv - xr[tid + j * 256]; dd += d * d; }
#pragma unroll
  for (int m = 16; m >= 1; m >>= 1) dd += __shfl_xor(dd, m, 32);
  if (lane == 0) red[wave] = dd;
  __syncthreads();
  if (tid == 0) {
    float t = 0.f;
#pragma unroll
    for (int k = 0; k < 8; ++k) t += red[k];
    atomicAdd(recon_acc, t);
  }
}

// ---------------- simple sum reduction (qloss mean numerator) ----------------
__global__ __launch_bounds__(256)
void reduce_sum_kernel(const float* __restrict__ v, int n, float* __restrict__ acc) {
  __shared__ float red[8];
  const int i = blockIdx.x * 256 + threadIdx.x;
  const int wave = threadIdx.x >> 5, lane = threadIdx.x & 31;
  float s = (i < n) ? v[i] : 0.f;
#pragma unroll
  for (int m = 16; m >= 1; m >>= 1) s += __shfl_xor(s, m, 32);
  if (lane == 0) red[wave] = s;
  __syncthreads();
  if (threadIdx.x == 0) {
    float t = 0.f;
#pragma unroll
    for (int k = 0; k < 8; ++k) t += red[k];
    atomicAdd(acc, t);
  }
}

// ---------------- pack (id0,id1,id2) into 48-bit keys ----------------
__global__ __launch_bounds__(256)
void pack_keys_kernel(const int* __restrict__ i0, const int* __restrict__ i1,
                      const int* __restrict__ i2, unsigned long long* __restrict__ keys, int n) {
  const int i = blockIdx.x * 256 + threadIdx.x;
  if (i < n)
    keys[i] = (unsigned long long)i0[i]
            | ((unsigned long long)i1[i] << 16)
            | ((unsigned long long)i2[i] << 32);
}

// ---------------- p_unique: rows with no duplicate at larger index ----------------
__global__ __launch_bounds__(256)
void unique_count_kernel(const unsigned long long* __restrict__ keys, int n,
                         int* __restrict__ cnt) {
  __shared__ unsigned long long tile[256];
  __shared__ int red[8];
  const int tid = threadIdx.x;
  const int i = blockIdx.x * 256 + tid;
  const unsigned long long k = keys[i];
  int dup = 0;
  const int ntiles = n / 256;
  for (int jt = blockIdx.x; jt < ntiles; ++jt) {   // only j >= block start matter
    tile[tid] = keys[jt * 256 + tid];
    __syncthreads();
    const int jbase = jt * 256;
    for (int jj = 0; jj < 256; ++jj) {             // LDS broadcast compare
      const int j = jbase + jj;
      dup |= (j > i) & (tile[jj] == k);
    }
    __syncthreads();
  }
  int u = dup ? 0 : 1;
#pragma unroll
  for (int m = 16; m >= 1; m >>= 1) u += __shfl_xor(u, m, 32);
  if ((tid & 31) == 0) red[tid >> 5] = u;
  __syncthreads();
  if (tid == 0) {
    int t = 0;
#pragma unroll
    for (int w = 0; w < 8; ++w) t += red[w];
    atomicAdd(cnt, t);
  }
}

// ---------------- init / finalize scalars ----------------
__global__ void init_scalars_kernel(float* r, float* q, int* u) {
  if (threadIdx.x == 0 && blockIdx.x == 0) { *r = 0.f; *q = 0.f; *u = 0; }
}
__global__ void finalize_kernel(const float* r, const float* q, const int* u,
                                float* out, int Bsz, int last) {
  if (threadIdx.x == 0 && blockIdx.x == 0) {
    const float recon = *r;
    const float qm = *q / (float)Bsz;
    out[0] = recon + qm;     // loss = mean(recon + qloss) = recon + mean(qloss)
    out[1] = recon;
    out[2] = qm;
    out[last] = (float)(*u) / (float)Bsz;
  }
}

// =====================================================================
// host side
// =====================================================================
extern "C" void kernel_launch(void* const* d_in, const int* in_sizes, int n_in,
                              void* d_out, int out_size, void* d_ws, size_t ws_size,
                              hipStream_t stream) {
  (void)n_in; (void)ws_size;
  const float* x   = (const float*)d_in[0];
  const float* ew[4] = {(const float*)d_in[1], (const float*)d_in[2],
                        (const float*)d_in[3], (const float*)d_in[4]};
  const float* eb[4] = {(const float*)d_in[5], (const float*)d_in[6],
                        (const float*)d_in[7], (const float*)d_in[8]};
  const float* dw[4] = {(const float*)d_in[9], (const float*)d_in[10],
                        (const float*)d_in[11], (const float*)d_in[12]};
  const float* db[4] = {(const float*)d_in[13], (const float*)d_in[14],
                        (const float*)d_in[15], (const float*)d_in[16]};
  const float* cb_emb = (const float*)d_in[17];
  const float* pw     = (const float*)d_in[18];
  const float* pb     = (const float*)d_in[19];
  const int B = in_sizes[0] / IN_DIM;   // 16384

  const int enc_dims[5] = {IN_DIM, 512, 256, 128, LATENT};
  const int dec_dims[5] = {LATENT, 128, 256, 512, IN_DIM};

  // workspace carve-up (buffers reused enc->dec)
  char* base = (char*)d_ws;
  size_t off = 0;
  auto carve = [&](size_t nbytes) -> char* {
    char* p = base + off;
    off = (off + nbytes + 255) & ~(size_t)255;
    return p;
  };
  float* xhat  = (float*)carve((size_t)B * IN_DIM * 4);
  float* h512  = (float*)carve((size_t)B * 512 * 4);
  float* h256  = (float*)carve((size_t)B * 256 * 4);
  float* h128  = (float*)carve((size_t)B * 128 * 4);
  float* res   = (float*)carve((size_t)B * LATENT * 4);
  float* esum  = (float*)carve((size_t)B * LATENT * 4);
  float* cbp   = (float*)carve((size_t)LQ * KCB * LATENT * 4);
  float* cnorm = (float*)carve((size_t)LQ * KCB * 4);
  float* qloss = (float*)carve((size_t)B * 4);
  int*   ids   = (int*)carve((size_t)LQ * B * 4);
  unsigned long long* keys = (unsigned long long*)carve((size_t)B * 8);
  __bf16* ewb[4];
  __bf16* dwb[4];
  for (int i = 0; i < 4; ++i)
    ewb[i] = (__bf16*)carve((size_t)enc_dims[i + 1] * enc_dims[i] * 2);
  for (int i = 0; i < 4; ++i)
    dwb[i] = (__bf16*)carve((size_t)dec_dims[i + 1] * dec_dims[i] * 2);
  __bf16* pwb = (__bf16*)carve((size_t)LQ * LATENT * LATENT * 2);
  __bf16* cbb = (__bf16*)carve((size_t)LQ * KCB * LATENT * 2);
  float* recon_acc = (float*)carve(4);
  float* qsum      = (float*)carve(4);
  int*   ucnt      = (int*)carve(4);

  init_scalars_kernel<<<1, 1, 0, stream>>>(recon_acc, qsum, ucnt);

  auto cvt = [&](const float* s, __bf16* d, size_t n) {
    const int n4 = (int)(n / 4);
    cvt_bf16_kernel<<<(n4 + 255) / 256, 256, 0, stream>>>(s, d, n4);
  };
  for (int i = 0; i < 4; ++i) cvt(ew[i], ewb[i], (size_t)enc_dims[i + 1] * enc_dims[i]);
  for (int i = 0; i < 4; ++i) cvt(dw[i], dwb[i], (size_t)dec_dims[i + 1] * dec_dims[i]);
  cvt(pw, pwb, (size_t)LQ * LATENT * LATENT);

  auto gemm = [&](const float* A, const __bf16* W, const float* bi, float* C,
                  int M, int N, int K, int relu) {
    gemm_bias_kernel<<<dim3(M / 128, N / 64), 256, 0, stream>>>(A, W, bi, C, M, N, K, relu);
  };

  // encoder (z written straight into residual buffer)
  gemm(x,    ewb[0], eb[0], h512, B, 512, IN_DIM, 1);
  gemm(h512, ewb[1], eb[1], h256, B, 256, 512, 1);
  gemm(h256, ewb[2], eb[2], h128, B, 128, 256, 1);
  gemm(h128, ewb[3], eb[3], res,  B, LATENT, 128, 0);

  // sim_vq codebook projection per layer: cb = cb_emb[l] @ proj_w[l]^T + proj_b[l]
  for (int l = 0; l < LQ; ++l)
    gemm(cb_emb + (size_t)l * KCB * LATENT, pwb + (size_t)l * LATENT * LATENT,
         pb + (size_t)l * LATENT, cbp + (size_t)l * KCB * LATENT,
         KCB, LATENT, LATENT, 0);
  rownorm2_kernel<<<(LQ * KCB * 32) / 256, 256, 0, stream>>>(cbp, cnorm, LQ * KCB);
  cvt(cbp, cbb, (size_t)LQ * KCB * LATENT);   // bf16 codebook for WMMA sweep

  // residual VQ loop (sequential across layers)
  for (int l = 0; l < LQ; ++l) {
    vq_argmin_kernel<<<B / 128, 256, 0, stream>>>(
        res, cbb + (size_t)l * KCB * LATENT, cnorm + (size_t)l * KCB, ids + (size_t)l * B);
    vq_update_kernel<<<(B * 32) / 256, 256, 0, stream>>>(
        cbp + (size_t)l * KCB * LATENT, ids + (size_t)l * B,
        res, esum, qloss, (float*)d_out + 3, l, (l == 0) ? 1 : 0, B);
  }

  // decoder
  gemm(esum, dwb[0], db[0], h128, B, 128, LATENT, 1);
  gemm(h128, dwb[1], db[1], h256, B, 256, 128, 1);
  gemm(h256, dwb[2], db[2], h512, B, 512, 256, 1);
  gemm(h512, dwb[3], db[3], xhat, B, IN_DIM, 512, 0);

  // losses + stats
  norm_recon_kernel<<<B, 256, 0, stream>>>(xhat, x, recon_acc);
  reduce_sum_kernel<<<B / 256, 256, 0, stream>>>(qloss, B, qsum);
  pack_keys_kernel<<<B / 256, 256, 0, stream>>>(ids, ids + B, ids + 2 * B, keys, B);
  unique_count_kernel<<<B / 256, 256, 0, stream>>>(keys, B, ucnt);
  finalize_kernel<<<1, 1, 0, stream>>>(recon_acc, qsum, ucnt, (float*)d_out, B, out_size - 1);
}
